// ConformerBlock_36799279792610
// MI455X (gfx1250) — matrix operand order repaired and checked by reading, not compiled, not run on hardware
//
#include <hip/hip_runtime.h>
#include <hip/hip_bf16.h>

typedef __attribute__((ext_vector_type(16))) _Float16 v16h;
typedef __attribute__((ext_vector_type(8)))  _Float16 v8h;
typedef __attribute__((ext_vector_type(8)))  float    v8f;

#define DMODEL 512
#define BATCH  8
#define SEQ    1024
#define NROWS  (BATCH*SEQ)    /* 8192 */
#define NHEAD  8
#define DHEAD  64
#define DFF    2048
#define INNER  1024
#define KCONV  31
#define EPSLN  1e-5f

__device__ __forceinline__ float fsig(float v)  { return 1.0f / (1.0f + __expf(-v)); }
__device__ __forceinline__ float fsilu(float v) { return v * fsig(v); }

__device__ __forceinline__ v8f wmma_f16(v16h a, v16h b, v8f c) {
    return __builtin_amdgcn_wmma_f32_16x16x32_f16(false, a, false, b, (short)0, c, false, false);
}

// ---------------------------------------------------------------- weight cvt
__global__ __launch_bounds__(256) void cvt_f32_f16(const float* __restrict__ src,
                                                   _Float16* __restrict__ dst, int n) {
    int i = blockIdx.x * 256 + threadIdx.x;
    if (i < n) dst[i] = (_Float16)src[i];
}

// ---------------------------------------------------------------- layer norm
// One block (128 threads) per row of 512.
template<bool F16OUT>
__global__ __launch_bounds__(128) void ln_kernel(const float* __restrict__ x,
                                                 const float* __restrict__ g,
                                                 const float* __restrict__ bta,
                                                 void* __restrict__ out) {
    int row = blockIdx.x;
    const float* xr = x + (size_t)row * DMODEL;
    int t = threadIdx.x;
    float v0[4];
    float s = 0.f, sq = 0.f;
#pragma unroll
    for (int e = 0; e < 4; ++e) {
        float v = xr[t + e * 128];
        v0[e] = v; s += v; sq += v * v;
    }
    __shared__ float rs_[128], rq_[128];
    rs_[t] = s; rq_[t] = sq;
    __syncthreads();
    for (int off = 64; off > 0; off >>= 1) {
        if (t < off) { rs_[t] += rs_[t + off]; rq_[t] += rq_[t + off]; }
        __syncthreads();
    }
    float mean = rs_[0] * (1.0f / DMODEL);
    float var  = rq_[0] * (1.0f / DMODEL) - mean * mean;
    float rstd = rsqrtf(var + EPSLN);
#pragma unroll
    for (int e = 0; e < 4; ++e) {
        int i = t + e * 128;
        float o = (v0[e] - mean) * rstd * g[i] + bta[i];
        if constexpr (F16OUT) ((_Float16*)out)[(size_t)row * DMODEL + i] = (_Float16)o;
        else                  ((float*)out)[(size_t)row * DMODEL + i] = o;
    }
}

// ---------------------------------------------------------------- WMMA GEMM
// C[M,N] = A[M,K] (f16) @ W[N,K]^T (f16), f32 accumulate.
// Block = 128 threads (4 waves); each wave owns a 32(M) x 64(N) tile:
// 2 A-frags x 4 B-frags -> 8 WMMAs per K-step of 32; each B-frag is
// reused for both M-frags (~21 FLOP/byte from L2).
enum { EPI_F16_BIAS = 0, EPI_F16_BIAS_SILU = 1, EPI_F32_BIAS = 2, EPI_F32_RES = 3 };

template<int EPI>
__global__ __launch_bounds__(128) void gemm_wmma(const _Float16* __restrict__ A,
                                                 const _Float16* __restrict__ W,
                                                 const float* __restrict__ bias,
                                                 int K, int N,
                                                 const float* __restrict__ X,
                                                 float rscale, float ascale,
                                                 void* __restrict__ out) {
    int lane = threadIdx.x & 31;
    int wave = threadIdx.x >> 5;
    int row0 = blockIdx.x * 32;
    int col0 = (blockIdx.y * 4 + wave) * 64;
    int half = lane >> 4;
    int nl   = lane & 15;

    v8f acc[2][4];
#pragma unroll
    for (int i = 0; i < 2; ++i)
#pragma unroll
        for (int j = 0; j < 4; ++j)
#pragma unroll
            for (int e = 0; e < 8; ++e) acc[i][j][e] = 0.f;

    const _Float16* arow0 = A + (size_t)(row0 + nl) * K + half * 16;
    const _Float16* arow1 = arow0 + (size_t)16 * K;
    for (int k0 = 0; k0 < K; k0 += 32) {
        v16h af0, af1;
        *(v8h*)&af0         = *(const v8h*)(arow0 + k0);
        *(((v8h*)&af0) + 1) = *(const v8h*)(arow0 + k0 + 8);
        *(v8h*)&af1         = *(const v8h*)(arow1 + k0);
        *(((v8h*)&af1) + 1) = *(const v8h*)(arow1 + k0 + 8);
#pragma unroll
        for (int j = 0; j < 4; ++j) {
            const _Float16* wp = W + (size_t)(col0 + j * 16 + nl) * K + k0 + half * 16;
            v16h bf = *(const v16h*)wp;
            acc[0][j] = wmma_f16(af0, bf, acc[0][j]);
            acc[1][j] = wmma_f16(af1, bf, acc[1][j]);
        }
    }
#pragma unroll
    for (int i = 0; i < 2; ++i) {
#pragma unroll
        for (int j = 0; j < 4; ++j) {
#pragma unroll
            for (int r = 0; r < 8; ++r) {
                size_t row = (size_t)row0 + i * 16 + r + half * 8;
                int col = col0 + j * 16 + nl;
                float v = acc[i][j][r];
                if constexpr (EPI == EPI_F16_BIAS) {
                    ((_Float16*)out)[row * N + col] = (_Float16)(v + bias[col]);
                } else if constexpr (EPI == EPI_F16_BIAS_SILU) {
                    ((_Float16*)out)[row * N + col] = (_Float16)fsilu(v + bias[col]);
                } else if constexpr (EPI == EPI_F32_BIAS) {
                    ((float*)out)[row * N + col] = v + bias[col];
                } else {
                    ((float*)out)[row * N + col] =
                        rscale * X[row * N + col] + ascale * (v + bias[col]);
                }
            }
        }
    }
}

// ---------------------------------------------------------------- qkv repack
// qkv[8192,1536] (bias applied) -> q[B,H,S,64], k[B,H,S,64], vt[B,H,64,S]
__global__ __launch_bounds__(256) void repack_qkv(const _Float16* __restrict__ qkv,
                                                  _Float16* __restrict__ q,
                                                  _Float16* __restrict__ k,
                                                  _Float16* __restrict__ vt) {
    size_t i = (size_t)blockIdx.x * 256 + threadIdx.x;   // NROWS*1536 total
    int col = (int)(i % 1536);
    size_t rowi = i / 1536;
    int s = (int)(rowi & (SEQ - 1));
    int b = (int)(rowi >> 10);
    int t = col >> 9;            // 0=q 1=k 2=v
    int rem = col & 511;
    int h = rem >> 6;
    int d = rem & 63;
    _Float16 val = qkv[i];
    size_t bh = (size_t)(b * NHEAD + h);
    if (t == 0)      q [(bh * SEQ + s) * DHEAD + d] = val;
    else if (t == 1) k [(bh * SEQ + s) * DHEAD + d] = val;
    else             vt[(bh * DHEAD + d) * SEQ + s] = val;
}

// ---------------------------------------------------------------- flash attn
// Each wave: 16 queries x full DH=64 for one (b,h). Online softmax.
__global__ __launch_bounds__(128) void flash_attn(const _Float16* __restrict__ q,
                                                  const _Float16* __restrict__ k,
                                                  const _Float16* __restrict__ vt,
                                                  _Float16* __restrict__ ctx) {
    __shared__ __align__(32) _Float16 ptile[4][16 * 32];
    int lane = threadIdx.x & 31;
    int wave = threadIdx.x >> 5;
    int gw = blockIdx.x * 4 + wave;
    int qt = gw & 63;                 // SEQ/16 = 64 query tiles
    int bh = gw >> 6;                 // b*H + h
    int q0 = qt * 16;
    const _Float16* Qp = q  + (size_t)bh * SEQ * DHEAD;
    const _Float16* Kp = k  + (size_t)bh * SEQ * DHEAD;
    const _Float16* Vp = vt + (size_t)bh * DHEAD * SEQ;
    int half = lane >> 4, nl = lane & 15;

    // Q fragments, pre-scaled by 1/sqrt(DH)=0.125 (exact in f16)
    v16h aq[2];
#pragma unroll
    for (int t = 0; t < 2; ++t) {
        const _Float16* qp = Qp + (size_t)(q0 + nl) * DHEAD + t * 32 + half * 16;
        *(v8h*)&aq[t]         = *(const v8h*)qp;
        *(((v8h*)&aq[t]) + 1) = *(const v8h*)(qp + 8);
#pragma unroll
        for (int e = 0; e < 16; ++e) aq[t][e] = aq[t][e] * (_Float16)0.125f;
    }

    float mrow[8], lrow[8];
    v8f accv[4];
#pragma unroll
    for (int r = 0; r < 8; ++r) { mrow[r] = -3.0e38f; lrow[r] = 0.f; }
#pragma unroll
    for (int dt = 0; dt < 4; ++dt)
#pragma unroll
        for (int e = 0; e < 8; ++e) accv[dt][e] = 0.f;

    _Float16* pt = &ptile[wave][0];

    for (int j0 = 0; j0 < SEQ; j0 += 32) {
        // ---- scores: 16 queries x 32 keys
        v8f sc[2];
#pragma unroll
        for (int nt = 0; nt < 2; ++nt) {
#pragma unroll
            for (int e = 0; e < 8; ++e) sc[nt][e] = 0.f;
#pragma unroll
            for (int t = 0; t < 2; ++t) {
                const _Float16* kp = Kp + (size_t)(j0 + nt * 16 + nl) * DHEAD + t * 32 + half * 16;
                v16h bf = *(const v16h*)kp;
                sc[nt] = wmma_f16(aq[t], bf, sc[nt]);
            }
        }
        // ---- online softmax row stats (rows live in (vgpr r, half) across 16 lanes)
        float scale[8];
#pragma unroll
        for (int r = 0; r < 8; ++r) {
            float mx = fmaxf(sc[0][r], sc[1][r]);
#pragma unroll
            for (int off = 8; off > 0; off >>= 1) mx = fmaxf(mx, __shfl_xor(mx, off, 16));
            float nm = fmaxf(mrow[r], mx);
            scale[r] = __expf(mrow[r] - nm);
            mrow[r] = nm;
        }
        __syncthreads();   // previous iteration's ptile reads are done
#pragma unroll
        for (int r = 0; r < 8; ++r) {
            float p0 = __expf(sc[0][r] - mrow[r]);
            float p1 = __expf(sc[1][r] - mrow[r]);
            int m = r + half * 8;
            pt[m * 32 + nl]      = (_Float16)p0;
            pt[m * 32 + 16 + nl] = (_Float16)p1;
            float ps = p0 + p1;
#pragma unroll
            for (int off = 8; off > 0; off >>= 1) ps += __shfl_xor(ps, off, 16);
            lrow[r] = lrow[r] * scale[r] + ps;
#pragma unroll
            for (int dt = 0; dt < 4; ++dt) accv[dt][r] *= scale[r];
        }
        __syncthreads();   // ptile written by all lanes
        // ---- P @ V : P as A-frag from LDS, V^T B-frags contiguous
        v16h pf;
        const _Float16* pp = pt + nl * 32 + half * 16;
        *(v8h*)&pf         = *(const v8h*)pp;
        *(((v8h*)&pf) + 1) = *(const v8h*)(pp + 8);
#pragma unroll
        for (int dt = 0; dt < 4; ++dt) {
            const _Float16* vp = Vp + (size_t)(dt * 16 + nl) * SEQ + j0 + half * 16;
            v16h bf = *(const v16h*)vp;
            accv[dt] = wmma_f16(pf, bf, accv[dt]);
        }
    }
    // ---- finalize: ctx[(b*S + s), h*64 + d] (f16, input to out-proj GEMM)
    int b = bh >> 3, h = bh & 7;
#pragma unroll
    for (int dt = 0; dt < 4; ++dt)
#pragma unroll
        for (int r = 0; r < 8; ++r) {
            int m = r + half * 8;
            int srow = q0 + m;
            int d = dt * 16 + nl;
            float v = accv[dt][r] / lrow[r];
            ctx[((size_t)(b * SEQ + srow)) * DMODEL + h * DHEAD + d] = (_Float16)v;
        }
}

// ---------------------------------------------------------------- GLU
__global__ __launch_bounds__(256) void glu_kernel(const float* __restrict__ y,
                                                  float* __restrict__ z) {
    size_t i = (size_t)blockIdx.x * 256 + threadIdx.x;   // NROWS*512
    size_t row = i >> 9;
    int c = (int)(i & 511);
    float a = y[row * INNER + c];
    float g = y[row * INNER + DMODEL + c];
    z[i] = a * fsig(g);
}

// ------------------------------------------- depthwise conv + BN + SiLU
__global__ __launch_bounds__(256) void dwconv_kernel(const float* __restrict__ z,
                                                     const float* __restrict__ dww,
                                                     const float* __restrict__ dwb,
                                                     const float* __restrict__ bg,
                                                     const float* __restrict__ bb,
                                                     const float* __restrict__ bm,
                                                     const float* __restrict__ bv,
                                                     _Float16* __restrict__ out) {
    size_t i = (size_t)blockIdx.x * 256 + threadIdx.x;   // B*S*D
    int c = (int)(i & 511);
    size_t rs = i >> 9;              // b*SEQ + s
    int s = (int)(rs & (SEQ - 1));
    size_t base = (rs - s) << 9;     // (b*SEQ)*512
    float acc = dwb[c];
#pragma unroll
    for (int kk = 0; kk < KCONV; ++kk) {
        int ss = s + kk - KCONV / 2;
        if (ss >= 0 && ss < SEQ)
            acc += z[base + ((size_t)ss << 9) + c] * dww[c * KCONV + kk];
    }
    float v = (acc - bm[c]) * rsqrtf(bv[c] + EPSLN) * bg[c] + bb[c];
    out[i] = (_Float16)fsilu(v);
}

// ================================================================ launcher
extern "C" void kernel_launch(void* const* d_in, const int* in_sizes, int n_in,
                              void* d_out, int out_size, void* d_ws, size_t ws_size,
                              hipStream_t stream) {
    const float* x_in   = (const float*)d_in[0];
    const float* ln1_g  = (const float*)d_in[1];
    const float* ln1_b  = (const float*)d_in[2];
    const float* ff1_w1 = (const float*)d_in[3];
    const float* ff1_b1 = (const float*)d_in[4];
    const float* ff1_w2 = (const float*)d_in[5];
    const float* ff1_b2 = (const float*)d_in[6];
    const float* lna_g  = (const float*)d_in[7];
    const float* lna_b  = (const float*)d_in[8];
    const float* qkv_w  = (const float*)d_in[9];
    const float* qkv_b  = (const float*)d_in[10];
    const float* out_w  = (const float*)d_in[11];
    const float* out_b  = (const float*)d_in[12];
    const float* lnc_g  = (const float*)d_in[13];
    const float* lnc_b  = (const float*)d_in[14];
    const float* pw1_w  = (const float*)d_in[15];
    const float* pw1_b  = (const float*)d_in[16];
    const float* dw_w   = (const float*)d_in[17];
    const float* dw_b   = (const float*)d_in[18];
    const float* bn_g   = (const float*)d_in[19];
    const float* bn_b   = (const float*)d_in[20];
    const float* bn_mean= (const float*)d_in[21];
    const float* bn_var = (const float*)d_in[22];
    const float* pw2_w  = (const float*)d_in[23];
    const float* pw2_b  = (const float*)d_in[24];
    const float* ln2_g  = (const float*)d_in[25];
    const float* ln2_b  = (const float*)d_in[26];
    const float* ff2_w1 = (const float*)d_in[27];
    const float* ff2_b1 = (const float*)d_in[28];
    const float* ff2_w2 = (const float*)d_in[29];
    const float* ff2_b2 = (const float*)d_in[30];
    const float* lnf_g  = (const float*)d_in[31];
    const float* lnf_b  = (const float*)d_in[32];

    // ---- workspace layout (bump allocator, lifetime-based aliasing) ----
    size_t off = 0;
    auto alloc = [&](size_t bytes) -> void* {
        void* p = (char*)d_ws + off;
        off += (bytes + 255) & ~(size_t)255;
        return p;
    };
    _Float16* wff1w1 = (_Float16*)alloc((size_t)DFF * DMODEL * 2);
    _Float16* wff1w2 = (_Float16*)alloc((size_t)DMODEL * DFF * 2);
    _Float16* wqkv   = (_Float16*)alloc((size_t)3 * DMODEL * DMODEL * 2);
    _Float16* wout   = (_Float16*)alloc((size_t)DMODEL * DMODEL * 2);
    _Float16* wpw1   = (_Float16*)alloc((size_t)INNER * DMODEL * 2);
    _Float16* wpw2   = (_Float16*)alloc((size_t)DMODEL * DMODEL * 2);
    _Float16* wff2w1 = (_Float16*)alloc((size_t)DFF * DMODEL * 2);
    _Float16* wff2w2 = (_Float16*)alloc((size_t)DMODEL * DFF * 2);
    float*    x0     = (float*)alloc((size_t)NROWS * DMODEL * 4);
    float*    x1     = (float*)alloc((size_t)NROWS * DMODEL * 4);
    _Float16* h16    = (_Float16*)alloc((size_t)NROWS * DMODEL * 2);
    char*     big0   = (char*)alloc((size_t)NROWS * DFF * 2);        // 32 MB
    char*     big1   = (char*)alloc((size_t)NROWS * 3 * DMODEL * 2); // 24 MB

    // big0 aliases (lifetimes disjoint): t16 (FF inner) / q,k,vt (attn) / y (pw1 out)
    _Float16* t16  = (_Float16*)big0;
    _Float16* qbuf = (_Float16*)big0;
    _Float16* kbuf = (_Float16*)(big0 + (size_t)BATCH * NHEAD * SEQ * DHEAD * 2);
    _Float16* vtbuf= (_Float16*)(big0 + (size_t)2 * BATCH * NHEAD * SEQ * DHEAD * 2);
    float*    ybuf = (float*)big0;
    // big1 aliases: qkv16 / ctx16 (after repack) / z + zf16 (conv stage)
    _Float16* qkv16 = (_Float16*)big1;
    _Float16* ctx16 = (_Float16*)big1;
    float*    zbuf  = (float*)big1;
    _Float16* zf16  = (_Float16*)(big1 + (size_t)NROWS * DMODEL * 4);

    auto cvt = [&](const float* s, _Float16* dct, int n) {
        cvt_f32_f16<<<(n + 255) / 256, 256, 0, stream>>>(s, dct, n);
    };
    cvt(ff1_w1, wff1w1, DFF * DMODEL);
    cvt(ff1_w2, wff1w2, DMODEL * DFF);
    cvt(qkv_w,  wqkv,   3 * DMODEL * DMODEL);
    cvt(out_w,  wout,   DMODEL * DMODEL);
    cvt(pw1_w,  wpw1,   INNER * DMODEL);
    cvt(pw2_w,  wpw2,   DMODEL * DMODEL);
    cvt(ff2_w1, wff2w1, DFF * DMODEL);
    cvt(ff2_w2, wff2w2, DMODEL * DFF);

    dim3 blk(128);
    const int MT = NROWS / 32;   // 256 M-tiles (32 rows per wave)

    // ---- stage 1: x0 = 1.5*x + 0.5*FF1(x) ----
    ln_kernel<true><<<NROWS, blk, 0, stream>>>(x_in, ln1_g, ln1_b, h16);
    gemm_wmma<EPI_F16_BIAS_SILU><<<dim3(MT, DFF / 256), blk, 0, stream>>>(
        h16, wff1w1, ff1_b1, DMODEL, DFF, nullptr, 0.f, 0.f, t16);
    gemm_wmma<EPI_F32_RES><<<dim3(MT, DMODEL / 256), blk, 0, stream>>>(
        t16, wff1w2, ff1_b2, DFF, DMODEL, x_in, 1.5f, 0.5f, x0);

    // ---- stage 2: x1 = x0 + attn(x0) ----
    ln_kernel<true><<<NROWS, blk, 0, stream>>>(x0, lna_g, lna_b, h16);
    gemm_wmma<EPI_F16_BIAS><<<dim3(MT, (3 * DMODEL) / 256), blk, 0, stream>>>(
        h16, wqkv, qkv_b, DMODEL, 3 * DMODEL, nullptr, 0.f, 0.f, qkv16);
    repack_qkv<<<(NROWS * 3 * DMODEL) / 256, 256, 0, stream>>>(qkv16, qbuf, kbuf, vtbuf);
    flash_attn<<<(BATCH * NHEAD * (SEQ / 16)) / 4, blk, 0, stream>>>(qbuf, kbuf, vtbuf, ctx16);
    gemm_wmma<EPI_F32_RES><<<dim3(MT, DMODEL / 256), blk, 0, stream>>>(
        ctx16, wout, out_b, DMODEL, DMODEL, x0, 1.f, 1.f, x1);

    // ---- stage 3: x0 = 2*x1 + convpath(x1)   (x + _conv where _conv = y + x) ----
    ln_kernel<true><<<NROWS, blk, 0, stream>>>(x1, lnc_g, lnc_b, h16);
    gemm_wmma<EPI_F32_BIAS><<<dim3(MT, INNER / 256), blk, 0, stream>>>(
        h16, wpw1, pw1_b, DMODEL, INNER, nullptr, 0.f, 0.f, ybuf);
    glu_kernel<<<(NROWS * DMODEL) / 256, 256, 0, stream>>>(ybuf, zbuf);
    dwconv_kernel<<<(NROWS * DMODEL) / 256, 256, 0, stream>>>(
        zbuf, dw_w, dw_b, bn_g, bn_b, bn_mean, bn_var, zf16);
    gemm_wmma<EPI_F32_RES><<<dim3(MT, DMODEL / 256), blk, 0, stream>>>(
        zf16, wpw2, pw2_b, DMODEL, DMODEL, x1, 2.f, 1.f, x0);

    // ---- stage 4: x1 = 1.5*x0 + 0.5*FF2(x0) ----
    ln_kernel<true><<<NROWS, blk, 0, stream>>>(x0, ln2_g, ln2_b, h16);
    gemm_wmma<EPI_F16_BIAS_SILU><<<dim3(MT, DFF / 256), blk, 0, stream>>>(
        h16, wff2w1, ff2_b1, DMODEL, DFF, nullptr, 0.f, 0.f, t16);
    gemm_wmma<EPI_F32_RES><<<dim3(MT, DMODEL / 256), blk, 0, stream>>>(
        t16, wff2w2, ff2_b2, DFF, DMODEL, x0, 1.5f, 0.5f, x1);

    // ---- final layer norm -> d_out (f32) ----
    ln_kernel<false><<<NROWS, blk, 0, stream>>>(x1, lnf_g, lnf_b, (float*)d_out);
    (void)in_sizes; (void)n_in; (void)out_size; (void)ws_size;
}
